// SwiGLUSGD_63170378990325
// MI455X (gfx1250) — matrix operand order repaired and checked by reading, not compiled, not run on hardware
//
#include <hip/hip_runtime.h>
#include <hip/hip_bf16.h>
#include <cstdint>

// ---------------------------------------------------------------------------
// Problem constants (match reference)
// ---------------------------------------------------------------------------
static constexpr int Bc  = 2;
static constexpr int Tc  = 2048;
static constexpr int Dc  = 1024;
static constexpr int Hc  = 8;
static constexpr int Kc  = 128;
static constexpr int Vc  = 128;
static constexpr int HIDc = 512;
static constexpr int BTc = 16;
static constexpr int NCc = 128;
static constexpr int NTOK = Bc * Tc;          // 4096
static constexpr int QKVG_LD = 4 * 1024;      // q|k|v|g packed columns

typedef __attribute__((ext_vector_type(16))) __bf16 v16bf;
typedef __attribute__((ext_vector_type(8)))  float  v8f;

// ---------------------------------------------------------------------------
// WMMA helpers (CDNA5 wave32, 16x16x32 bf16 -> f32)
// ---------------------------------------------------------------------------
__device__ __forceinline__ v8f wmma_bf16(v16bf a, v16bf b, v8f c) {
  return __builtin_amdgcn_wmma_f32_16x16x32_bf16(false, a, false, b, (short)0, c,
                                                 false, false);
}

__device__ __forceinline__ v8f zero_v8f() {
  return v8f{0.f, 0.f, 0.f, 0.f, 0.f, 0.f, 0.f, 0.f};
}

// A-fragment, source row-major [M][Kd]: lane row=r0+(l&15), elems 0..15 =
// K kb..kb+15, kb = (l>>4)*16. One 32B read.
__device__ __forceinline__ v16bf frag_a_rm(const __bf16* s, int ld, int r0,
                                           int k0, int lane) {
  const __bf16* p = s + (size_t)(r0 + (lane & 15)) * ld + k0 + ((lane >> 4) << 4);
  v16bf f;
#pragma unroll
  for (int i = 0; i < 8; ++i) { f[i] = p[i]; f[8 + i] = p[8 + i]; }
  return f;
}

// A-fragment from transposed source [Kdim][M] with zero-padding for k>=kvalid.
__device__ __forceinline__ v16bf frag_a_tr(const __bf16* s, int ld, int r0,
                                           int k0, int lane, int kvalid) {
  int r = r0 + (lane & 15);
  int kb = k0 + ((lane >> 4) << 4);
  v16bf f;
#pragma unroll
  for (int i = 0; i < 8; ++i) {
    int k1 = kb + i, k2 = kb + 8 + i;
    f[i]     = (k1 < kvalid) ? s[(size_t)k1 * ld + r] : (__bf16)0.0f;
    f[8 + i] = (k2 < kvalid) ? s[(size_t)k2 * ld + r] : (__bf16)0.0f;
  }
  return f;
}

// B-fragment, N-major source [N][Kd] (K contiguous): lane col n=n0+(l&15),
// elems 0..15 = K kb..kb+15, kb = (l>>4)*16. One 32B read.
__device__ __forceinline__ v16bf frag_b_nm(const __bf16* bt, int ld, int n0,
                                           int k0, int lane) {
  const __bf16* p = bt + (size_t)(n0 + (lane & 15)) * ld + k0 + ((lane >> 4) << 4);
  v16bf f;
#pragma unroll
  for (int i = 0; i < 16; ++i) f[i] = p[i];
  return f;
}

// B-fragment from K-major source [Kdim][N] (strided), zero-padded k>=kvalid.
__device__ __forceinline__ v16bf frag_b_km(const __bf16* b, int ld, int n0,
                                           int k0, int lane, int kvalid) {
  int n = n0 + (lane & 15);
  int kb = k0 + ((lane >> 4) << 4);
  v16bf f;
#pragma unroll
  for (int i = 0; i < 16; ++i) {
    int kk = kb + i;
    f[i] = (kk < kvalid) ? b[(size_t)kk * ld + n] : (__bf16)0.0f;
  }
  return f;
}

__device__ __forceinline__ float sigf(float x) { return 1.0f / (1.0f + __expf(-x)); }

// ---------------------------------------------------------------------------
// Conversion / packing kernels
// ---------------------------------------------------------------------------
__global__ void cvt_f32_bf16(const float* __restrict__ src,
                             __bf16* __restrict__ dst, size_t n) {
  for (size_t i = (size_t)blockIdx.x * blockDim.x + threadIdx.x; i < n;
       i += (size_t)gridDim.x * blockDim.x)
    dst[i] = (__bf16)src[i];
}

// Pack Wq|Wk|Wv|Wg (each [D][1024], row-major) transposed into Wt [4096][D] bf16.
__global__ void pack_qkvg_wt(const float* __restrict__ Wq,
                             const float* __restrict__ Wk,
                             const float* __restrict__ Wv,
                             const float* __restrict__ Wg,
                             __bf16* __restrict__ Wt) {
  size_t n = (size_t)4096 * Dc;
  for (size_t i = (size_t)blockIdx.x * blockDim.x + threadIdx.x; i < n;
       i += (size_t)gridDim.x * blockDim.x) {
    int col = (int)(i >> 10);          // 0..4095 output row (= original column)
    int d   = (int)(i & 1023);         // reduction dim
    const float* W = (col < 1024) ? Wq : (col < 2048) ? Wk : (col < 3072) ? Wv : Wg;
    Wt[i] = (__bf16)W[(size_t)d * 1024 + (col & 1023)];
  }
}

// Transpose W_o [HV][D] -> Wot [D][HV] bf16.
__global__ void transpose_wo(const float* __restrict__ Wo, __bf16* __restrict__ Wot) {
  size_t n = (size_t)Dc * (Hc * Vc);
  for (size_t i = (size_t)blockIdx.x * blockDim.x + threadIdx.x; i < n;
       i += (size_t)gridDim.x * blockDim.x) {
    int nn = (int)(i >> 10);           // output row (= W_o column, 0..1023)
    int k  = (int)(i & 1023);
    Wot[i] = (__bf16)Wo[(size_t)k * Dc + nn];
  }
}

// theta = BASE_LR * sigmoid(x @ W_theta), stored [NTOK][H] f32.
__global__ void theta_proj(const float* __restrict__ x,
                           const float* __restrict__ Wth,
                           float* __restrict__ theta) {
  int idx = blockIdx.x * blockDim.x + threadIdx.x;
  if (idx >= NTOK * Hc) return;
  int tok = idx >> 3, h = idx & 7;
  const float* xr = x + (size_t)tok * Dc;
  float acc = 0.0f;
  for (int d = 0; d < Dc; ++d) acc += xr[d] * Wth[d * Hc + h];
  theta[idx] = 0.1f * sigf(acc);
}

// ---------------------------------------------------------------------------
// Tiled WMMA GEMM: C[M][N] (f32) = A[M][Kd] (bf16, row-major) @ Bt[N][Kd]^T
// Block tile 128x256, 256 threads (8 waves, 2x4), wave tile 64x64.
// Double-buffered LDS staging: one barrier per 32-K slab, staging of slab k+1
// overlaps WMMAs of slab k.
// ---------------------------------------------------------------------------
__global__ __launch_bounds__(256) void gemm_bf16_nt(
    const __bf16* __restrict__ A, const __bf16* __restrict__ Bt,
    float* __restrict__ C, int M, int N, int Kd) {
  __shared__ __bf16 As[2 * 128 * 32];
  __shared__ __bf16 Bs[2 * 256 * 32];

  const int tid = threadIdx.x, lane = tid & 31, wave = tid >> 5;
  const int bm = blockIdx.x * 128, bn = blockIdx.y * 256;
  const int wm = (wave >> 2) * 64;          // 0 / 64
  const int wn = (wave & 3) * 64;           // 0/64/128/192

  const int arow = tid >> 1;                // 0..127
  const int asub = (tid & 1) << 4;          // 0 / 16 elements

  v8f acc[4][4];
#pragma unroll
  for (int i = 0; i < 4; ++i)
#pragma unroll
    for (int j = 0; j < 4; ++j) acc[i][j] = zero_v8f();

  auto stage = [&](int buf, int k0) {
    const uint4* ga = (const uint4*)(A + (size_t)(bm + arow) * Kd + k0 + asub);
    uint4* sa = (uint4*)(As + buf * (128 * 32) + arow * 32 + asub);
    sa[0] = ga[0];
    sa[1] = ga[1];
    const uint4* gb = (const uint4*)(Bt + (size_t)(bn + tid) * Kd + k0);
    uint4* sb = (uint4*)(Bs + buf * (256 * 32) + tid * 32);
#pragma unroll
    for (int i = 0; i < 4; ++i) sb[i] = gb[i];
  };

  stage(0, 0);
  int buf = 0;
  for (int k0 = 0; k0 < Kd; k0 += 32) {
    if (k0 + 32 < Kd) {  // emits global_prefetch_b8 for the slab after next
      __builtin_prefetch(A + (size_t)(bm + arow) * Kd + k0 + 64 + asub, 0, 1);
      __builtin_prefetch(Bt + (size_t)(bn + tid) * Kd + k0 + 64, 0, 1);
    }
    __syncthreads();                        // slab `buf` fully staged
    if (k0 + 32 < Kd) stage(buf ^ 1, k0 + 32);

    const __bf16* as0 = As + buf * (128 * 32);
    const __bf16* bs0 = Bs + buf * (256 * 32);
    v16bf af[4], bfm[4];
#pragma unroll
    for (int i = 0; i < 4; ++i) af[i] = frag_a_rm(as0, 32, wm + 16 * i, 0, lane);
#pragma unroll
    for (int j = 0; j < 4; ++j) bfm[j] = frag_b_nm(bs0, 32, wn + 16 * j, 0, lane);
#pragma unroll
    for (int i = 0; i < 4; ++i)
#pragma unroll
      for (int j = 0; j < 4; ++j) acc[i][j] = wmma_bf16(af[i], bfm[j], acc[i][j]);
    buf ^= 1;
  }

#pragma unroll
  for (int i = 0; i < 4; ++i)
#pragma unroll
    for (int j = 0; j < 4; ++j) {
      int cn = bn + wn + 16 * j + (lane & 15);
      int cr = bm + wm + 16 * i + ((lane >> 4) << 3);
#pragma unroll
      for (int e = 0; e < 8; ++e) C[(size_t)(cr + e) * N + cn] = acc[i][j][e];
    }
}

// ---------------------------------------------------------------------------
// Scan kernel: one block per (b,h); fast-weight state resident in LDS (bf16).
// 512 threads = 16 waves. 128 sequential chunk steps, 7 WMMA matmul stages.
// ---------------------------------------------------------------------------
static constexpr size_t SCAN_LDS_BYTES =
    (size_t)(HIDc * Kc + Vc * HIDc + BTc * HIDc * 2 + BTc * Kc * 4) * 2 + 64;

__global__ __launch_bounds__(512) void swiglu_sgd_scan(
    const float* __restrict__ qkvg,    // [NTOK][4096] f32 : q|k|v|g
    const float* __restrict__ theta,   // [NTOK][H] f32
    const float* __restrict__ w1_init, // [B,H,HID,K] f32
    const float* __restrict__ w2_init, // [B,H,V,HID] f32
    float* __restrict__ outputs) {     // [NTOK][H][V] f32
  extern __shared__ char smem[];
  __bf16* w1s = (__bf16*)smem;                 // [HID][K]   (N-major for z/zo)
  __bf16* w2s = w1s + HIDc * Kc;               // [V][HID]   (N-major for pred/out)
  __bf16* zs  = w2s + Vc * HIDc;               // [BT][HID]  z, later dz
  __bf16* as_ = zs + BTc * HIDc;               // [BT][HID]  silu(z), later silu(zo)
  __bf16* kcs = as_ + BTc * HIDc;              // [BT][K]
  __bf16* vcs = kcs + BTc * Kc;                // [BT][V]
  __bf16* qcs = vcs + BTc * Vc;                // [BT][K]
  __bf16* tes = qcs + BTc * Kc;                // [BT][V]
  float*  ths = (float*)(tes + BTc * Vc);      // [BT]

  const int tid = threadIdx.x, lane = tid & 31, wave = tid >> 5;
  const int b = blockIdx.x >> 3, h = blockIdx.x & 7;

  // --- init fast weights (f32 -> bf16) ---
  {
    const float* w1g = w1_init + (size_t)(b * Hc + h) * HIDc * Kc;
    const float* w2g = w2_init + (size_t)(b * Hc + h) * Vc * HIDc;
    for (int i = tid; i < HIDc * Kc; i += 512) w1s[i] = (__bf16)w1g[i];
    for (int i = tid; i < Vc * HIDc; i += 512) w2s[i] = (__bf16)w2g[i];
  }

  for (int c = 0; c < NCc; ++c) {
    const int t0 = c * BTc;
    __syncthreads();

    // --- load chunk q/k/v (+theta) ---
    for (int i = tid; i < BTc * Kc; i += 512) {
      int t = i >> 7, x = i & 127;
      size_t base = (size_t)(b * Tc + t0 + t) * QKVG_LD + h * Kc + x;
      qcs[i] = (__bf16)qkvg[base];
      kcs[i] = (__bf16)qkvg[base + 1024];
      vcs[i] = (__bf16)qkvg[base + 2048];
    }
    if (tid < BTc) ths[tid] = theta[(size_t)(b * Tc + t0 + tid) * Hc + h];
    __syncthreads();

    // --- stage 1: z = kc @ w1^T ; a = silu(z). 32 hid tiles / 16 waves ---
    {
      v16bf ka[4];
#pragma unroll
      for (int kk = 0; kk < 4; ++kk) ka[kk] = frag_a_rm(kcs, Kc, 0, kk * 32, lane);
#pragma unroll
      for (int u = 0; u < 2; ++u) {
        int n0 = (wave * 2 + u) * 16;
        v8f acc = zero_v8f();
#pragma unroll
        for (int kk = 0; kk < 4; ++kk)
          acc = wmma_bf16(ka[kk], frag_b_nm(w1s, Kc, n0, kk * 32, lane), acc);
        int hid = n0 + (lane & 15), tb = (lane >> 4) << 3;
#pragma unroll
        for (int e = 0; e < 8; ++e) {
          float zv = acc[e];
          zs[(tb + e) * HIDc + hid]  = (__bf16)zv;
          as_[(tb + e) * HIDc + hid] = (__bf16)(zv * sigf(zv));
        }
      }
    }
    __syncthreads();

    // --- stage 2: pred = a @ w2^T ; te = theta*(pred - v). 8 v-tiles ---
    if (wave < 8) {
      int n0 = wave * 16;
      v8f acc = zero_v8f();
#pragma unroll
      for (int kk = 0; kk < HIDc; kk += 32)
        acc = wmma_bf16(frag_a_rm(as_, HIDc, 0, kk, lane),
                        frag_b_nm(w2s, HIDc, n0, kk, lane), acc);
      int v = n0 + (lane & 15), tb = (lane >> 4) << 3;
#pragma unroll
      for (int e = 0; e < 8; ++e) {
        int t = tb + e;
        float err = acc[e] - (float)vcs[t * Vc + v];
        tes[t * Vc + v] = (__bf16)(ths[t] * err);
      }
    }
    __syncthreads();

    // --- stage 3: da = te @ w2 ; dz = da * dsilu(z) (overwrite zs) ---
    {
      v16bf ta[4];
#pragma unroll
      for (int kk = 0; kk < 4; ++kk) ta[kk] = frag_a_rm(tes, Vc, 0, kk * 32, lane);
#pragma unroll
      for (int u = 0; u < 2; ++u) {
        int n0 = (wave * 2 + u) * 16;
        v8f acc = zero_v8f();
#pragma unroll
        for (int kk = 0; kk < 4; ++kk)
          acc = wmma_bf16(ta[kk], frag_b_km(w2s, HIDc, n0, kk * 32, lane, Vc), acc);
        int hid = n0 + (lane & 15), tb = (lane >> 4) << 3;
#pragma unroll
        for (int e = 0; e < 8; ++e) {
          int t = tb + e;
          float zv = (float)zs[t * HIDc + hid];
          float s = sigf(zv);
          float dsz = s * (1.0f + zv * (1.0f - s));
          zs[t * HIDc + hid] = (__bf16)(acc[e] * dsz);
        }
      }
    }
    __syncthreads();

    // --- stage 4: w2 -= te^T @ a. Wave w: v-tile w>>1, hid tiles (w&1)*16+j.
    //     A-fragment (te^T, K=16 zero-padded) is loaded ONCE per wave. ---
    {
      int m0 = (wave >> 1) * 16;
      v16bf afrag = frag_a_tr(tes, Vc, m0, 0, lane, BTc);
      int vb = m0 + ((lane >> 4) << 3);
#pragma unroll
      for (int j = 0; j < 16; ++j) {
        int n0 = ((wave & 1) * 16 + j) * 16;
        v8f acc = wmma_bf16(afrag, frag_b_km(as_, HIDc, n0, 0, lane, BTc),
                            zero_v8f());
        int hid = n0 + (lane & 15);
#pragma unroll
        for (int e = 0; e < 8; ++e) {
          int off = (vb + e) * HIDc + hid;
          w2s[off] = (__bf16)((float)w2s[off] - acc[e]);
        }
      }
    }
    // --- stage 5: w1 -= dz^T @ kc. 8 kc B-fragments hoisted to registers,
    //     2 hid-tile A-fragments per wave. ---
    {
      v16bf bfrag[8];
#pragma unroll
      for (int j = 0; j < 8; ++j)
        bfrag[j] = frag_b_km(kcs, Kc, j * 16, 0, lane, BTc);
#pragma unroll
      for (int u = 0; u < 2; ++u) {
        int m0 = (wave * 2 + u) * 16;
        v16bf afrag = frag_a_tr(zs, HIDc, m0, 0, lane, BTc);
        int hb = m0 + ((lane >> 4) << 3);
#pragma unroll
        for (int j = 0; j < 8; ++j) {
          v8f acc = wmma_bf16(afrag, bfrag[j], zero_v8f());
          int kx = j * 16 + (lane & 15);
#pragma unroll
          for (int e = 0; e < 8; ++e) {
            int off = (hb + e) * Kc + kx;
            w1s[off] = (__bf16)((float)w1s[off] - acc[e]);
          }
        }
      }
    }
    __syncthreads();

    // --- stage 6: zo = qc @ w1n^T ; as_ = silu(zo) ---
    {
      v16bf qa[4];
#pragma unroll
      for (int kk = 0; kk < 4; ++kk) qa[kk] = frag_a_rm(qcs, Kc, 0, kk * 32, lane);
#pragma unroll
      for (int u = 0; u < 2; ++u) {
        int n0 = (wave * 2 + u) * 16;
        v8f acc = zero_v8f();
#pragma unroll
        for (int kk = 0; kk < 4; ++kk)
          acc = wmma_bf16(qa[kk], frag_b_nm(w1s, Kc, n0, kk * 32, lane), acc);
        int hid = n0 + (lane & 15), tb = (lane >> 4) << 3;
#pragma unroll
        for (int e = 0; e < 8; ++e) {
          float zv = acc[e];
          as_[(tb + e) * HIDc + hid] = (__bf16)(zv * sigf(zv));
        }
      }
    }
    __syncthreads();

    // --- stage 7: out = silu(zo) @ w2n^T -> global [tok][h][v] f32 ---
    if (wave < 8) {
      int n0 = wave * 16;
      v8f acc = zero_v8f();
#pragma unroll
      for (int kk = 0; kk < HIDc; kk += 32)
        acc = wmma_bf16(frag_a_rm(as_, HIDc, 0, kk, lane),
                        frag_b_nm(w2s, HIDc, n0, kk, lane), acc);
      int v = n0 + (lane & 15), tb = (lane >> 4) << 3;
#pragma unroll
      for (int e = 0; e < 8; ++e)
        outputs[((size_t)(b * Tc + t0 + tb + e) * Hc + h) * Vc + v] = acc[e];
    }
  }
}

// ---------------------------------------------------------------------------
// Gated RMSNorm: o * rsqrt(mean(o^2)+eps) * norm_w * silu(g) -> bf16 [NTOK][HV]
// One block (128 threads) per (tok, h) row.
// ---------------------------------------------------------------------------
__global__ __launch_bounds__(128) void rmsnorm_gate(
    const float* __restrict__ outs,    // [NTOK*H][V]
    const float* __restrict__ qkvg,    // g at column offset 3072
    const float* __restrict__ norm_w,  // [V]
    __bf16* __restrict__ onb) {        // [NTOK][H*V]
  int row = blockIdx.x;                // (tok*H + h)
  int v = threadIdx.x;
  float o = outs[(size_t)row * Vc + v];
  __shared__ float red[128];
  red[v] = o * o;
  __syncthreads();
  for (int s = 64; s > 0; s >>= 1) {
    if (v < s) red[v] += red[v + s];
    __syncthreads();
  }
  float r = rsqrtf(red[0] * (1.0f / Vc) + 1e-5f);
  size_t tok = (size_t)(row >> 3);
  int h = row & 7;
  float g = qkvg[tok * QKVG_LD + 3072 + h * Vc + v];
  float val = o * r * norm_w[v] * (g * sigf(g));
  onb[tok * (Hc * Vc) + h * Vc + v] = (__bf16)val;
}

// ---------------------------------------------------------------------------
// Host-side orchestration
// ---------------------------------------------------------------------------
extern "C" void kernel_launch(void* const* d_in, const int* in_sizes, int n_in,
                              void* d_out, int out_size, void* d_ws, size_t ws_size,
                              hipStream_t stream) {
  const float* x   = (const float*)d_in[0];
  const float* Wq  = (const float*)d_in[1];
  const float* Wk  = (const float*)d_in[2];
  const float* Wv  = (const float*)d_in[3];
  const float* Wth = (const float*)d_in[4];
  const float* Wg  = (const float*)d_in[5];
  const float* nw  = (const float*)d_in[6];
  const float* Wo  = (const float*)d_in[7];
  const float* w1i = (const float*)d_in[8];
  const float* w2i = (const float*)d_in[9];
  float* out = (float*)d_out;

  char* p = (char*)d_ws;
  auto alloc = [&](size_t bytes) -> char* {
    char* r = p;
    p += (bytes + 255) & ~(size_t)255;
    return r;
  };
  __bf16* xb   = (__bf16*)alloc((size_t)NTOK * Dc * 2);        // inputs bf16
  __bf16* Wt   = (__bf16*)alloc((size_t)4096 * Dc * 2);        // packed qkvg Wt
  __bf16* Wot  = (__bf16*)alloc((size_t)Dc * (Hc * Vc) * 2);   // W_o transposed
  float*  Call = (float*) alloc((size_t)NTOK * QKVG_LD * 4);   // q|k|v|g f32
  float*  th   = (float*) alloc((size_t)NTOK * Hc * 4);        // theta
  float*  outs = (float*) alloc((size_t)NTOK * Hc * Vc * 4);   // scan outputs
  __bf16* onb  = (__bf16*)alloc((size_t)NTOK * (Hc * Vc) * 2); // normed, bf16

  // 1) precision conversion / packing
  cvt_f32_bf16<<<2048, 256, 0, stream>>>(x, xb, (size_t)NTOK * Dc);
  pack_qkvg_wt<<<2048, 256, 0, stream>>>(Wq, Wk, Wv, Wg, Wt);
  transpose_wo<<<1024, 256, 0, stream>>>(Wo, Wot);

  // 2) theta projection (small)
  theta_proj<<<(NTOK * Hc + 255) / 256, 256, 0, stream>>>(x, Wth, th);

  // 3) fused qkvg projection: [4096 x 1024] @ [1024 x 4096]
  gemm_bf16_nt<<<dim3(NTOK / 128, QKVG_LD / 256), 256, 0, stream>>>(
      xb, Wt, Call, NTOK, QKVG_LD, Dc);

  // 4) sequential fast-weight scan: one WG per (b,h), state in LDS
  swiglu_sgd_scan<<<Bc * Hc, 512, SCAN_LDS_BYTES, stream>>>(Call, th, w1i, w2i,
                                                            outs);

  // 5) gated RMSNorm -> bf16
  rmsnorm_gate<<<NTOK * Hc, 128, 0, stream>>>(outs, Call, nw, onb);

  // 6) output projection: [4096 x 1024] @ [1024 x 1024] -> d_out (f32)
  gemm_bf16_nt<<<dim3(NTOK / 128, Dc / 256), 256, 0, stream>>>(
      onb, Wot, out, NTOK, Dc, Dc);
}
